// EMESC_16535624089617
// MI455X (gfx1250) — matrix operand order repaired and checked by reading, not compiled, not run on hardware
//
#include <hip/hip_runtime.h>
#include <hip/hip_bf16.h>
#include <cstdint>

typedef _Float16 v16h __attribute__((ext_vector_type(16)));
typedef _Float16 v8h  __attribute__((ext_vector_type(8)));
typedef float    v8f  __attribute__((ext_vector_type(8)));

#define DEV __device__ __forceinline__

constexpr int B_SZ    = 4;
constexpr int HW      = 144;
constexpr int N_PIX   = B_SZ * HW * HW;     // 82944 (multiple of 16)
constexpr int N_SLK   = N_PIX + 16;         // slack rows for 3x3 window over-read
constexpr int DSZ     = 300;
constexpr int K_PAD   = 320;                // K padded to 10 tiles of 32
constexpr int K_TILES = 10;
constexpr int C_TILES = 19;                 // 19*16 = 304 >= 300 output cols
constexpr int ROW_TILES = N_PIX / 16;       // 5184
constexpr int OW      = 142;
constexpr int N_OUT   = B_SZ * OW * OW;     // 80656 (multiple of 16)
constexpr int OUT_TILES = N_OUT / 16;       // 5041
constexpr int HID2    = 128;

DEV float gelu_erf(float x) { return 0.5f * x * (1.0f + erff(x * 0.70710678118654752f)); }
DEV float sigm(float x)     { return 1.0f / (1.0f + expf(-x)); }

DEV v8f wmma_f16(v8f acc, v16h a, v16h b) {
  // D = A(16x32 f16) * B(32x16 f16) + C(16x16 f32)
  return __builtin_amdgcn_wmma_f32_16x16x32_f16(false, a, false, b, (short)0, acc, false, false);
}

// Build A fragment (16x32 f16) for this lane from a row-major [16 x K_stride] tile.
// lane<16: row=lane,    K = k0+[0..7]  and k0+[16..23]
// lane>=16: row=lane-16, K = k0+[8..15] and k0+[24..31]
DEV v16h make_a(const _Float16* p) {
  v8h lo = *(const v8h*)p;
  v8h hi = *(const v8h*)(p + 16);
  v16h a;
#pragma unroll
  for (int i = 0; i < 8; ++i) { a[i] = lo[i]; a[i + 8] = hi[i]; }
  return a;
}

// Async global->LDS copy of one 16-byte chunk (CDNA5 GLOBAL_LOAD_ASYNC_TO_LDS_B128,
// tracked by ASYNCcnt; bypasses VGPRs).
DEV void async_g2l_b128(uint32_t lds_byte_off, const void* gaddr) {
  asm volatile("global_load_async_to_lds_b128 %0, %1, off"
               :: "v"(lds_byte_off), "v"((uint64_t)(uintptr_t)gaddr)
               : "memory");
}
DEV void wait_async0() { asm volatile("s_wait_asynccnt 0x0" ::: "memory"); }

// ---------------------------------------------------------------------------
// Weight repack: w[out=300][in=300] f32 -> B-fragment-ordered f16
// layout: [jt(19)][kt(10)][lane(32)][16 halves], lane n<16 -> col jt*16+n, K=kt*32+{0..7,16..23}[r]
//         lane n>=16 -> col jt*16+n-16, K=kt*32+{8..15,24..31}[r]
// ---------------------------------------------------------------------------
__global__ __launch_bounds__(256) void k_pack300(const float* __restrict__ w,
                                                 _Float16* __restrict__ out) {
  int idx = blockIdx.x * 256 + threadIdx.x;
  if (idx >= C_TILES * K_TILES * 32 * 16) return;
  int r = idx & 15;
  int lane = (idx >> 4) & 31;
  int kt = (idx >> 9) % K_TILES;
  int jt = (idx >> 9) / K_TILES;
  int col = jt * 16 + (lane & 15);
  int k = kt * 32 + (lane >> 4) * 8 + (r & 7) + (r >> 3) * 16;
  float v = (col < DSZ && k < DSZ) ? w[col * DSZ + k] : 0.0f;
  out[idx] = (_Float16)v;
}

// conv weight: w1[128][300][3][3] -> [tap(9)][jt(8)][kt(10)][lane(32)][16]
__global__ __launch_bounds__(256) void k_packconv(const float* __restrict__ w,
                                                   _Float16* __restrict__ out) {
  int idx = blockIdx.x * 256 + threadIdx.x;
  if (idx >= 9 * 8 * K_TILES * 32 * 16) return;
  int r = idx & 15;
  int lane = (idx >> 4) & 31;
  int t = idx >> 9;
  int kt = t % K_TILES;
  int jt = (t / K_TILES) % 8;
  int tap = t / (K_TILES * 8);
  int di = tap / 3, dj = tap % 3;
  int col = jt * 16 + (lane & 15);                // out channel (0..127)
  int k = kt * 32 + (lane >> 4) * 8 + (r & 7) + (r >> 3) * 16;  // in channel
  float v = (k < DSZ) ? w[((size_t)(col * DSZ + k) * 3 + di) * 3 + dj] : 0.0f;
  out[idx] = (_Float16)v;
}

// fc2 weight: w2[128][128] -> [jt(8)][kt(4)][lane(32)][16]
__global__ __launch_bounds__(256) void k_packfc2(const float* __restrict__ w,
                                                  _Float16* __restrict__ out) {
  int idx = blockIdx.x * 256 + threadIdx.x;
  if (idx >= 8 * 4 * 32 * 16) return;
  int r = idx & 15;
  int lane = (idx >> 4) & 31;
  int t = idx >> 9;
  int kt = t % 4;
  int jt = t / 4;
  int col = jt * 16 + (lane & 15);
  int k = kt * 32 + (lane >> 4) * 8 + (r & 7) + (r >> 3) * 16;
  out[idx] = (_Float16)w[col * HID2 + k];
}

// ---------------------------------------------------------------------------
// Encoder: six 21->300 gconvs + initial cell state
// ---------------------------------------------------------------------------
__global__ __launch_bounds__(256) void k_encoder(
    const float* __restrict__ y,
    const float* __restrict__ we1w, const float* __restrict__ we1b,
    const float* __restrict__ we2w, const float* __restrict__ we2b,
    const float* __restrict__ wf1w, const float* __restrict__ wf1b,
    const float* __restrict__ wi1w, const float* __restrict__ wi1b,
    const float* __restrict__ wf2w, const float* __restrict__ wf2b,
    const float* __restrict__ wi2w, const float* __restrict__ wi2b,
    _Float16* __restrict__ ywe1, _Float16* __restrict__ ywe2,
    _Float16* __restrict__ ywf1, _Float16* __restrict__ ywi1,
    _Float16* __restrict__ ywf2, _Float16* __restrict__ ywi2,
    float* __restrict__ Ca, _Float16* __restrict__ Xa, _Float16* __restrict__ Xb) {
  size_t idx = (size_t)blockIdx.x * 256 + threadIdx.x;
  if (idx >= (size_t)N_PIX * K_PAD) return;
  int d = (int)(idx % K_PAD);
  size_t pix = idx / K_PAD;
  if (d >= DSZ) {  // zero K-padding of both ping-pong matmul inputs
    Xa[pix * K_PAD + d] = (_Float16)0.0f;
    Xb[pix * K_PAD + d] = (_Float16)0.0f;
    return;
  }
  int b = (int)(pix / (HW * HW));
  int rem = (int)(pix % (HW * HW));
  float yv[21];
#pragma unroll
  for (int c = 0; c < 21; ++c) yv[c] = y[((size_t)(b * 21 + c)) * (HW * HW) + rem];
  float s0 = we1b[d], s1 = we2b[d], s2 = wf1b[d], s3 = wi1b[d], s4 = wf2b[d], s5 = wi2b[d];
#pragma unroll
  for (int c = 0; c < 21; ++c) {
    float v = yv[c];
    s0 += v * we1w[d * 21 + c];
    s1 += v * we2w[d * 21 + c];
    s2 += v * wf1w[d * 21 + c];
    s3 += v * wi1w[d * 21 + c];
    s4 += v * wf2w[d * 21 + c];
    s5 += v * wi2w[d * 21 + c];
  }
  float ge1 = gelu_erf(s0), ge2 = gelu_erf(s1), gf1 = gelu_erf(s2);
  float gi1 = gelu_erf(s3), gf2 = gelu_erf(s4), gi2 = gelu_erf(s5);
  size_t p = pix * DSZ + d;
  ywe1[p] = (_Float16)ge1; ywe2[p] = (_Float16)ge2;
  ywf1[p] = (_Float16)gf1; ywi1[p] = (_Float16)gi1;
  ywf2[p] = (_Float16)gf2; ywi2[p] = (_Float16)gi2;
  float c12 = sigm(gi1) * (-ge1);
  Ca[p] = c12;
  Xa[pix * K_PAD + d] = (_Float16)fmaxf(c12, 0.0f);
}

// ---------------------------------------------------------------------------
// Fused half-step: three WMMA matmuls (Wd/Wf/Wi) + gating
//   ct = Cin*sigm(gelu(X@Wf+bf)+ywF) + (ywE + Add - gelu(X@Wd+bd))*sigm(gelu(X@Wi+bi)+ywI)
//   Cout = ct ; Xout = relu(ct) as f16
// NOTE: Add and Xout may alias (step2) -> no __restrict__ on them.
// ---------------------------------------------------------------------------
__global__ __launch_bounds__(256) void k_half(
    const _Float16* __restrict__ Xin, const float* __restrict__ Cin,
    const _Float16* Add,
    const _Float16* __restrict__ ywE, const _Float16* __restrict__ ywF,
    const _Float16* __restrict__ ywI,
    const _Float16* __restrict__ Wd, const _Float16* __restrict__ Wf,
    const _Float16* __restrict__ Wi,
    const float* __restrict__ bd, const float* __restrict__ bf,
    const float* __restrict__ bi,
    float* __restrict__ Cout, _Float16* Xout) {
  __shared__ _Float16 lx[16 * K_PAD];  // 10 KB A tile
  const int tid = threadIdx.x;
  const int rt = blockIdx.x;  // row tile (16 pixels)
  // Async global->LDS stage of the 16x320 f16 A tile (640 x b128 chunks).
  {
    const char* src = (const char*)(Xin + (size_t)rt * 16 * K_PAD);
    uint32_t lbase = (uint32_t)(uintptr_t)lx;
#pragma unroll
    for (int it = 0; it < 3; ++it) {
      int idx = tid + it * 256;
      if (idx < (16 * K_PAD * 2) / 16)
        async_g2l_b128(lbase + idx * 16, src + (size_t)idx * 16);
    }
    wait_async0();
  }
  __syncthreads();
  const int wave = tid >> 5, lane = tid & 31;
  const int halfsel = lane >> 4, ml = lane & 15;

  for (int jt = wave; jt < C_TILES; jt += 8) {  // wave-uniform -> EXEC stays full
    v8f ad = {0.f, 0.f, 0.f, 0.f, 0.f, 0.f, 0.f, 0.f};
    v8f af = ad, ai = ad;
    const size_t bbase = ((size_t)jt * K_TILES * 32 + lane) * 16;
    // Double-buffered B fragments: loads for kt+1 issue under kt's WMMAs.
    v16h bdc = *(const v16h*)(Wd + bbase);
    v16h bfc = *(const v16h*)(Wf + bbase);
    v16h bic = *(const v16h*)(Wi + bbase);
#pragma unroll
    for (int kt = 0; kt < K_TILES; ++kt) {
      const int k0 = kt * 32;
      v16h a = make_a(&lx[ml * K_PAD + k0 + halfsel * 8]);
      v16h bdn = bdc, bfn = bfc, bin = bic;
      if (kt + 1 < K_TILES) {
        const size_t nb = bbase + (size_t)(kt + 1) * 512;
        bdn = *(const v16h*)(Wd + nb);
        bfn = *(const v16h*)(Wf + nb);
        bin = *(const v16h*)(Wi + nb);
      }
      ad = wmma_f16(ad, a, bdc);
      af = wmma_f16(af, a, bfc);
      ai = wmma_f16(ai, a, bic);
      bdc = bdn; bfc = bfn; bic = bin;
    }
    const int col = jt * 16 + ml;
    const int rbase = rt * 16 + halfsel * 8;  // C/D: vgpr i -> M = i + 8*(lane>=16)
    if (col < DSZ) {
      float vbd = bd[col], vbf = bf[col], vbi = bi[col];
#pragma unroll
      for (int i = 0; i < 8; ++i) {
        size_t row = (size_t)(rbase + i);
        size_t p = row * DSZ + col;
        float gd = gelu_erf(ad[i] + vbd);
        float gf = gelu_erf(af[i] + vbf);
        float gi = gelu_erf(ai[i] + vbi);
        float add = Add ? (float)Add[row * K_PAD + col] : 0.0f;
        float ctl = (float)ywE[p] + add - gd;
        float ft = sigm(gf + (float)ywF[p]);
        float it = sigm(gi + (float)ywI[p]);
        float ct = Cin[p] * ft + ctl * it;
        Cout[p] = ct;
        Xout[row * K_PAD + col] = (_Float16)fmaxf(ct, 0.0f);
      }
    } else {  // cols 300..303: keep matmul-input padding zero
#pragma unroll
      for (int i = 0; i < 8; ++i)
        Xout[(size_t)(rbase + i) * K_PAD + col] = (_Float16)0.0f;
    }
  }
}

// ---------------------------------------------------------------------------
// 3x3 valid conv D->128 as 9 accumulated WMMAs + erf-GELU
// Block = one output row (b,i); waves own out-channel tiles; loop pixel tiles.
// ---------------------------------------------------------------------------
__global__ __launch_bounds__(256) void k_conv3x3(const _Float16* __restrict__ X,
                                                  const _Float16* __restrict__ Wc,
                                                  const float* __restrict__ b1,
                                                  _Float16* __restrict__ H) {
  __shared__ _Float16 ls[3 * 18 * K_PAD];  // 3 rows x 18 pixels x 320ch = 33.75 KB
  const int blk = blockIdx.x;
  const int b = blk / OW, i = blk % OW;
  const int tid = threadIdx.x, wave = tid >> 5, lane = tid & 31;
  const int halfsel = lane >> 4, ml = lane & 15;
  const int ROWD = 18 * K_PAD / 2;  // dwords per di-row chunk

  for (int pt = 0; pt < 9; ++pt) {  // 9 pixel tiles cover j=0..143 (142 valid)
    __syncthreads();
    const int j0 = pt * 16;
    for (int q = tid; q < 3 * ROWD; q += 256) {
      int di = q / ROWD, d2 = q % ROWD;
      size_t srcPix = (size_t)(b * HW + i + di) * HW + j0;  // may spill into slack rows
      ((uint32_t*)ls)[di * ROWD + d2] = ((const uint32_t*)X)[srcPix * (K_PAD / 2) + d2];
    }
    __syncthreads();
    v8f acc = {0.f, 0.f, 0.f, 0.f, 0.f, 0.f, 0.f, 0.f};
    // Flattened (tap,kt) loop with double-buffered B fragments.
    auto boff = [&](int t) -> size_t {
      int tap = t / K_TILES, kt = t % K_TILES;
      return (((size_t)tap * 8 * K_TILES + wave * K_TILES + kt) * 32 + lane) * 16;
    };
    v16h bc = *(const v16h*)(Wc + boff(0));
#pragma unroll
    for (int t = 0; t < 9 * K_TILES; ++t) {
      const int tap = t / K_TILES, kt = t % K_TILES;
      const int di = tap / 3, dj = tap % 3;
      v16h a = make_a(&ls[((di * 18) + (ml + dj)) * K_PAD + kt * 32 + halfsel * 8]);
      v16h bn = bc;
      if (t + 1 < 9 * K_TILES) bn = *(const v16h*)(Wc + boff(t + 1));
      acc = wmma_f16(acc, a, bc);
      bc = bn;
    }
    const int col = wave * 16 + ml;  // out channel
    const float bias = b1[col];
#pragma unroll
    for (int ii = 0; ii < 8; ++ii) {
      const int j = j0 + halfsel * 8 + ii;
      if (j < OW) {
        size_t op = (size_t)(b * OW + i) * OW + j;
        H[op * HID2 + col] = (_Float16)gelu_erf(acc[ii] + bias);
      }
    }
  }
}

// 1x1 128->128 + GELU
__global__ __launch_bounds__(256) void k_fc2(const _Float16* __restrict__ H,
                                              const _Float16* __restrict__ W2,
                                              const float* __restrict__ b2,
                                              _Float16* __restrict__ H2) {
  __shared__ _Float16 lh[16 * HID2];
  const int rt = blockIdx.x, tid = threadIdx.x;
  {
    const uint32_t* g = (const uint32_t*)(H + (size_t)rt * 16 * HID2);
#pragma unroll
    for (int q = 0; q < (16 * HID2 / 2) / 256; ++q)
      ((uint32_t*)lh)[tid + q * 256] = g[tid + q * 256];
  }
  __syncthreads();
  const int wave = tid >> 5, lane = tid & 31;
  const int halfsel = lane >> 4, ml = lane & 15;
  v8f acc = {0.f, 0.f, 0.f, 0.f, 0.f, 0.f, 0.f, 0.f};
#pragma unroll
  for (int kt = 0; kt < 4; ++kt) {
    v16h a = make_a(&lh[ml * HID2 + kt * 32 + halfsel * 8]);
    v16h bfr = *(const v16h*)(W2 + (((size_t)wave * 4 + kt) * 32 + lane) * 16);
    acc = wmma_f16(acc, a, bfr);
  }
  const int col = wave * 16 + ml;
  const float bias = b2[col];
#pragma unroll
  for (int ii = 0; ii < 8; ++ii) {
    size_t row = (size_t)rt * 16 + halfsel * 8 + ii;
    H2[row * HID2 + col] = (_Float16)gelu_erf(acc[ii] + bias);
  }
}

// 1x1 128->1 (no activation)
__global__ __launch_bounds__(256) void k_fc3(const _Float16* __restrict__ H2,
                                              const float* __restrict__ w3,
                                              const float* __restrict__ b3,
                                              float* __restrict__ out) {
  int p = blockIdx.x * 256 + threadIdx.x;
  if (p >= N_OUT) return;
  float s = b3[0];
#pragma unroll 16
  for (int c = 0; c < HID2; ++c) s += (float)H2[(size_t)p * HID2 + c] * w3[c];
  out[p] = s;
}

// ---------------------------------------------------------------------------
extern "C" void kernel_launch(void* const* d_in, const int* in_sizes, int n_in,
                              void* d_out, int out_size, void* d_ws, size_t ws_size,
                              hipStream_t stream) {
  (void)in_sizes; (void)n_in; (void)out_size; (void)ws_size;
  const float* y    = (const float*)d_in[0];
  const float* we1w = (const float*)d_in[1];  const float* we1b = (const float*)d_in[2];
  const float* we2w = (const float*)d_in[3];  const float* we2b = (const float*)d_in[4];
  const float* wf1w = (const float*)d_in[5];  const float* wf1b = (const float*)d_in[6];
  const float* wi1w = (const float*)d_in[7];  const float* wi1b = (const float*)d_in[8];
  const float* wf2w = (const float*)d_in[9];  const float* wf2b = (const float*)d_in[10];
  const float* wi2w = (const float*)d_in[11]; const float* wi2b = (const float*)d_in[12];
  const float* wd1w = (const float*)d_in[13]; const float* wd1b = (const float*)d_in[14];
  const float* wd2w = (const float*)d_in[15]; const float* wd2b = (const float*)d_in[16];
  const float* wfx1w= (const float*)d_in[17]; const float* wfx1b= (const float*)d_in[18];
  const float* wix1w= (const float*)d_in[19]; const float* wix1b= (const float*)d_in[20];
  const float* wfx2w= (const float*)d_in[21]; const float* wfx2b= (const float*)d_in[22];
  const float* wix2w= (const float*)d_in[23]; const float* wix2b= (const float*)d_in[24];
  const float* lw1  = (const float*)d_in[25]; const float* lb1  = (const float*)d_in[26];
  const float* lw2  = (const float*)d_in[27]; const float* lb2  = (const float*)d_in[28];
  const float* lw3  = (const float*)d_in[29]; const float* lb3  = (const float*)d_in[30];

  char* ws = (char*)d_ws;
  size_t off = 0;
  auto alloc = [&](size_t bytes) -> char* {
    char* r = ws + off;
    off = (off + bytes + 255) & ~(size_t)255;
    return r;
  };
  _Float16* yw[6];
  for (int m = 0; m < 6; ++m) yw[m] = (_Float16*)alloc((size_t)N_PIX * DSZ * 2);
  // order: e1, e2, f1, i1, f2, i2
  float*    Ca = (float*)alloc((size_t)N_PIX * DSZ * 4);
  float*    Cb = (float*)alloc((size_t)N_PIX * DSZ * 4);
  _Float16* Xa = (_Float16*)alloc((size_t)N_SLK * K_PAD * 2);
  _Float16* Xb = (_Float16*)alloc((size_t)N_SLK * K_PAD * 2);
  const size_t PKB = (size_t)C_TILES * K_TILES * 512;
  _Float16* Pd1 = (_Float16*)alloc(PKB * 2); _Float16* Pd2 = (_Float16*)alloc(PKB * 2);
  _Float16* Pf1 = (_Float16*)alloc(PKB * 2); _Float16* Pi1 = (_Float16*)alloc(PKB * 2);
  _Float16* Pf2 = (_Float16*)alloc(PKB * 2); _Float16* Pi2 = (_Float16*)alloc(PKB * 2);
  _Float16* Pc  = (_Float16*)alloc((size_t)9 * 8 * K_TILES * 512 * 2);
  _Float16* P2  = (_Float16*)alloc((size_t)8 * 4 * 512 * 2);
  _Float16* Hb  = (_Float16*)alloc((size_t)N_OUT * HID2 * 2);
  _Float16* H2b = (_Float16*)alloc((size_t)N_OUT * HID2 * 2);

  const int pkGrid = (int)((PKB + 255) / 256);
  k_pack300<<<pkGrid, 256, 0, stream>>>(wd1w, Pd1);
  k_pack300<<<pkGrid, 256, 0, stream>>>(wd2w, Pd2);
  k_pack300<<<pkGrid, 256, 0, stream>>>(wfx1w, Pf1);
  k_pack300<<<pkGrid, 256, 0, stream>>>(wix1w, Pi1);
  k_pack300<<<pkGrid, 256, 0, stream>>>(wfx2w, Pf2);
  k_pack300<<<pkGrid, 256, 0, stream>>>(wix2w, Pi2);
  k_packconv<<<(9 * 8 * K_TILES * 512 + 255) / 256, 256, 0, stream>>>(lw1, Pc);
  k_packfc2<<<(8 * 4 * 512 + 255) / 256, 256, 0, stream>>>(lw2, P2);

  k_encoder<<<(int)(((size_t)N_PIX * K_PAD) / 256), 256, 0, stream>>>(
      y, we1w, we1b, we2w, we2b, wf1w, wf1b, wi1w, wi1b, wf2w, wf2b, wi2w, wi2b,
      yw[0], yw[1], yw[2], yw[3], yw[4], yw[5], Ca, Xa, Xb);

  // step2 uses the '2' gates (ywe2,ywfy2,ywiy2, wd2,wfx2,wix2); body uses '1'.
  auto step2 = [&](const _Float16* add) {
    k_half<<<ROW_TILES, 256, 0, stream>>>(Xa, Ca, add, yw[1], yw[4], yw[5],
                                          Pd2, Pf2, Pi2, wd2b, wfx2b, wix2b, Cb, Xb);
  };
  auto body = [&]() {
    k_half<<<ROW_TILES, 256, 0, stream>>>(Xb, Cb, Xb, yw[0], yw[2], yw[3],
                                          Pd1, Pf1, Pi1, wd1b, wfx1b, wix1b, Ca, Xa);
  };
  step2(nullptr);                       // xt1 = 0
  for (int t = 0; t < 3; ++t) { body(); step2(Xb); }
  // final xt1 (relu(ct1), f16, K-padded) lives in Xb

  k_conv3x3<<<B_SZ * OW, 256, 0, stream>>>(Xb, Pc, lb1, Hb);
  k_fc2<<<OUT_TILES, 256, 0, stream>>>(Hb, P2, lb2, H2b);
  k_fc3<<<(N_OUT + 255) / 256, 256, 0, stream>>>(H2b, lw3, lb3, (float*)d_out);
}
